// Seq2Seq_57243324121294
// MI455X (gfx1250) — compile-verified, matrix operand order checked by reading, not run on hardware
//
#include <hip/hip_runtime.h>
#include <cstdint>

// ---------------- model dims ----------------
#define LSRC 128
#define BB   32
#define TT   64
#define T1   63          // T-1 decode steps
#define EE   512
#define HH   512
#define GG   1536        // 3*H
#define VV   32000

typedef __attribute__((ext_vector_type(16))) _Float16 v16h;
typedef __attribute__((ext_vector_type(8)))  float    v8f;
typedef __attribute__((ext_vector_type(4)))  float    f32x4;

// ---------------- WMMA helpers (gfx1250 wave32) ----------------
// Load 16 halves of one A/B fragment slice: [p..p+7] and [p+16..p+23]
__device__ __forceinline__ v16h load_frag16(const _Float16* p) {
    union { f32x4 q[2]; v16h h; } u;
    u.q[0] = *reinterpret_cast<const f32x4*>(p);
    u.q[1] = *reinterpret_cast<const f32x4*>(p + 16);
    return u.h;
}

__device__ __forceinline__ v8f wmma_f16(v16h a, v16h b, v8f c) {
    return __builtin_amdgcn_wmma_f32_16x16x32_f16(false, a, false, b, (short)0, c, false, false);
}

// One 16x16 output tile of  C += A[m0:m0+16, :K] * W[n0:n0+16, :K]^T
// A row-major (lda halves), W row-major (ldb halves).  K % 32 == 0.
__device__ __forceinline__ v8f tile_gemm(const _Float16* A, int lda,
                                         const _Float16* W, int ldb,
                                         int m0, int n0, int K, int lane, v8f acc) {
    const int koff = (lane >> 4) << 3;           // 0 or 8
    const _Float16* pa = A + (size_t)(m0 + (lane & 15)) * lda + koff;
    const _Float16* pb = W + (size_t)(n0 + (lane & 15)) * ldb + koff;
    for (int k = 0; k < K; k += 32)
        acc = wmma_f16(load_frag16(pa + k), load_frag16(pb + k), acc);
    return acc;
}

// ---------------- blocked tiled GEMM:  C = act(A * W^T + bias) ----------------
// A:[M,K] f16 row-major, W:[N,K] f16 row-major.  Each wave computes a 32x64
// output block (2 M-tiles x 4 N-tiles, 8 f32 accumulators): per k-step, 6
// fragment loads feed 8 WMMAs (2.7x better L2 intensity than 1-tile/wave).
// Waves within a block share the N-block so the streamed W rows hit cache.
// Requires M%32==0, N%64==0, K%32==0.
template <int OUTF16, int RELU>
__global__ void gemm_nt(const _Float16* __restrict__ A, const _Float16* __restrict__ W,
                        const float* __restrict__ bias, void* __restrict__ Cout,
                        int M, int N, int K, int lda, int ldb, int ldc) {
    const int lane = threadIdx.x & 31;
    const int wave = threadIdx.x >> 5;
    const int tile = blockIdx.x * (blockDim.x >> 5) + wave;
    const int nrows = M >> 5;                    // 32-row blocks
    const int ncols = N >> 6;                    // 64-col blocks
    if (tile >= nrows * ncols) return;           // wave-uniform guard
    const int m0 = (tile % nrows) << 5;          // waves in a block share n-block
    const int n0 = (tile / nrows) << 6;

    const int koff = (lane >> 4) << 3;           // 0 or 8
    const _Float16* pa0 = A + (size_t)(m0 + (lane & 15)) * lda + koff;
    const _Float16* pa1 = pa0 + (size_t)16 * lda;
    const _Float16* pb0 = W + (size_t)(n0 + (lane & 15)) * ldb + koff;
    const _Float16* pb1 = pb0 + (size_t)16 * ldb;
    const _Float16* pb2 = pb0 + (size_t)32 * ldb;
    const _Float16* pb3 = pb0 + (size_t)48 * ldb;

    v8f acc[2][4];
    for (int mi = 0; mi < 2; ++mi)
        for (int ni = 0; ni < 4; ++ni) acc[mi][ni] = (v8f){};

    for (int k = 0; k < K; k += 32) {
        __builtin_prefetch(pb0 + k + 512, 0, 1);   // stream W 16 k-steps (512 halves) ahead
        v16h a0 = load_frag16(pa0 + k);
        v16h a1 = load_frag16(pa1 + k);
        v16h b0 = load_frag16(pb0 + k);
        v16h b1 = load_frag16(pb1 + k);
        v16h b2 = load_frag16(pb2 + k);
        v16h b3 = load_frag16(pb3 + k);
        acc[0][0] = wmma_f16(a0, b0, acc[0][0]);
        acc[0][1] = wmma_f16(a0, b1, acc[0][1]);
        acc[0][2] = wmma_f16(a0, b2, acc[0][2]);
        acc[0][3] = wmma_f16(a0, b3, acc[0][3]);
        acc[1][0] = wmma_f16(a1, b0, acc[1][0]);
        acc[1][1] = wmma_f16(a1, b1, acc[1][1]);
        acc[1][2] = wmma_f16(a1, b2, acc[1][2]);
        acc[1][3] = wmma_f16(a1, b3, acc[1][3]);
    }

    for (int mi = 0; mi < 2; ++mi) {
        const int mb = m0 + mi * 16 + ((lane >> 4) << 3);
        for (int ni = 0; ni < 4; ++ni) {
            const int n = n0 + ni * 16 + (lane & 15);
            const float bv = bias ? bias[n] : 0.0f;
            for (int r = 0; r < 8; ++r) {
                float v = acc[mi][ni][r] + bv;
                if (RELU) v = fmaxf(v, 0.0f);
                if (OUTF16) ((_Float16*)Cout)[(size_t)(mb + r) * ldc + n] = (_Float16)v;
                else        ((float*)Cout)[(size_t)(mb + r) * ldc + n]    = v;
            }
        }
    }
}

// ---------------- small utility kernels ----------------
__global__ void cvt_f32_f16(const float* __restrict__ s, _Float16* __restrict__ d, int n) {
    int i = blockIdx.x * blockDim.x + threadIdx.x;
    if (i < n) d[i] = (_Float16)s[i];
}

__global__ void gather_enc(const int* __restrict__ tok, const float* __restrict__ emb,
                           _Float16* __restrict__ out) {
    int r = blockIdx.x;                           // r = l*32 + b
    const float* s = emb + (size_t)tok[r] * EE;
    _Float16* d = out + (size_t)r * EE;
    for (int j = threadIdx.x; j < EE; j += blockDim.x) d[j] = (_Float16)s[j];
}

__global__ void gather_dec(const int* __restrict__ trgt, const float* __restrict__ emb,
                           _Float16* __restrict__ out) {
    int r = blockIdx.x;                           // r = t*32 + b, t < 63
    int t = r >> 5, b = r & 31;
    const float* s = emb + (size_t)trgt[b * TT + t] * EE;
    _Float16* d = out + (size_t)r * EE;
    for (int j = threadIdx.x; j < EE; j += blockDim.x) d[j] = (_Float16)s[j];
}

// enc_sum[b*128+l][j] = y1[l*32+b][j] + y1[l*32+b][512+j]   (transpose L,B -> B,L)
__global__ void enc_sum_tr(const _Float16* __restrict__ y1, _Float16* __restrict__ out) {
    int r = blockIdx.x;                           // r = b*128 + l
    int b = r >> 7, l = r & 127;
    const _Float16* s = y1 + (size_t)(l * 32 + b) * 1024;
    _Float16* d = out + (size_t)r * HH;
    for (int j = threadIdx.x; j < HH; j += blockDim.x)
        d[j] = (_Float16)((float)s[j] + (float)s[HH + j]);
}

__device__ __forceinline__ float sigm(float x) { return 1.0f / (1.0f + __expf(-x)); }

// ---------------- persistent encoder GRU (one block per direction) ----------------
// gi:[2][L*B][G] f32 (x@Wih^T + bih precomputed), Whh:[2][G][H] f16, bhh:[2][G] f32
// ys:[L*B][1024] f16, this direction writes column half dir*512.
__global__ void __launch_bounds__(1024) gru_enc(const float* __restrict__ gi,
                                                const _Float16* __restrict__ Whh,
                                                const float* __restrict__ bhh,
                                                _Float16* __restrict__ ys) {
    extern __shared__ char smem[];
    _Float16* hF = (_Float16*)smem;               // [32][512]
    float*    gh = (float*)(smem + 32 * HH * 2);  // [32][1536]
    const int dir = blockIdx.x;
    gi  += (size_t)dir * LSRC * BB * GG;
    Whh += (size_t)dir * GG * HH;
    bhh += dir * GG;
    const int tid = threadIdx.x, lane = tid & 31, wave = tid >> 5;
    float hreg[16];
    for (int i = 0; i < 16; ++i) { hreg[i] = 0.f; hF[tid + 1024 * i] = (_Float16)0.f; }
    __syncthreads();
    for (int t = 0; t < LSRC; ++t) {
        const int ta = dir ? (LSRC - 1 - t) : t;
        // gh = hF @ Whh^T + bhh  (192 tiles / 32 waves = 6 each)
        for (int tt = 0; tt < 6; ++tt) {
            const int tile = wave * 6 + tt;
            const int m0 = (tile / 96) << 4, n0 = (tile % 96) << 4;
            v8f acc = {};
            acc = tile_gemm(hF, HH, Whh, HH, m0, n0, HH, lane, acc);
            const int n = n0 + (lane & 15), mb = m0 + ((lane >> 4) << 3);
            const float bv = bhh[n];
            for (int r = 0; r < 8; ++r) gh[(mb + r) * GG + n] = acc[r] + bv;
        }
        __syncthreads();
        const float* gir = gi + (size_t)(ta * BB) * GG;
        for (int i = 0; i < 16; ++i) {
            const int pos = tid + 1024 * i;       // = b*512 + j
            const int b = pos >> 9, j = pos & 511;
            const float* g  = gir + b * GG;
            const float* gg = gh + b * GG;
            float r  = sigm(g[j] + gg[j]);
            float z  = sigm(g[HH + j] + gg[HH + j]);
            float nn = tanhf(g[2 * HH + j] + r * gg[2 * HH + j]);
            float hn = (1.f - z) * nn + z * hreg[i];
            hreg[i] = hn;
            hF[pos] = (_Float16)hn;
            ys[(size_t)(ta * BB + b) * 1024 + dir * HH + j] = (_Float16)hn;
        }
        __syncthreads();
    }
}

// ---------------- persistent decoder: GRU x2 + attention, emits feat ----------------
__global__ void __launch_bounds__(1024) gru_dec(
    const float* __restrict__ gi_emb,      // [63*32][1536] rows t*32+b (incl bih0)
    const _Float16* __restrict__ WihCtx,   // dec_Wih0_f16 + 512, row stride 1024
    const _Float16* __restrict__ Whh0, const float* __restrict__ bhh0,
    const _Float16* __restrict__ Wih1, const float* __restrict__ bih1,
    const _Float16* __restrict__ Whh1, const float* __restrict__ bhh1,
    const _Float16* __restrict__ Wd,   const float* __restrict__ bd,
    const _Float16* __restrict__ encq,     // [32*128][512]
    const _Float16* __restrict__ encs,     // [32*128][512]
    _Float16* __restrict__ feat) {         // [2016][1024], row b*63+t
    extern __shared__ char smem[];
    _Float16* gIH  = (_Float16*)smem;          // [32][1536]
    _Float16* gHH  = gIH + 32 * GG;            // [32][1536]
    _Float16* h0F  = gHH + 32 * GG;            // [32][512]
    _Float16* h1F  = h0F + 32 * HH;
    _Float16* ctxF = h1F + 32 * HH;
    _Float16* qF   = gIH;                      // alias (after gates consumed)
    float*    ebuf = (float*)gHH;              // alias, [32][128]
    const int tid = threadIdx.x, lane = tid & 31, wave = tid >> 5;
    float h0r[16], h1r[16];
    for (int i = 0; i < 16; ++i) {
        h0r[i] = 0.f; h1r[i] = 0.f;
        h0F[tid + 1024 * i] = (_Float16)0.f;
        h1F[tid + 1024 * i] = (_Float16)0.f;
        ctxF[tid + 1024 * i] = (_Float16)0.f;
    }
    __syncthreads();
    for (int t = 0; t < T1; ++t) {
        const float* ge = gi_emb + (size_t)(t * BB) * GG;
        // stage gIH = gi_emb + ctx@WihCtx^T ; gHH = h0@Whh0^T + bhh0
        for (int tt = 0; tt < 6; ++tt) {
            const int tile = wave * 6 + tt;
            const int m0 = (tile / 96) << 4, n0 = (tile % 96) << 4;
            const int n = n0 + (lane & 15), mb = m0 + ((lane >> 4) << 3);
            v8f a1 = {}; a1 = tile_gemm(ctxF, HH, WihCtx, 1024, m0, n0, HH, lane, a1);
            for (int r = 0; r < 8; ++r)
                gIH[(mb + r) * GG + n] = (_Float16)(a1[r] + ge[(size_t)(mb + r) * GG + n]);
            v8f a2 = {}; a2 = tile_gemm(h0F, HH, Whh0, HH, m0, n0, HH, lane, a2);
            const float bv = bhh0[n];
            for (int r = 0; r < 8; ++r) gHH[(mb + r) * GG + n] = (_Float16)(a2[r] + bv);
        }
        __syncthreads();
        for (int i = 0; i < 16; ++i) {          // h0 update
            const int pos = tid + 1024 * i, b = pos >> 9, j = pos & 511;
            const _Float16* gA = gIH + b * GG;
            const _Float16* gB = gHH + b * GG;
            float r  = sigm((float)gA[j] + (float)gB[j]);
            float z  = sigm((float)gA[HH + j] + (float)gB[HH + j]);
            float nn = tanhf((float)gA[2 * HH + j] + r * (float)gB[2 * HH + j]);
            float hn = (1.f - z) * nn + z * h0r[i];
            h0r[i] = hn; h0F[pos] = (_Float16)hn;
        }
        __syncthreads();
        // stage gIH = h0n@Wih1^T + bih1 ; gHH = h1@Whh1^T + bhh1
        for (int tt = 0; tt < 6; ++tt) {
            const int tile = wave * 6 + tt;
            const int m0 = (tile / 96) << 4, n0 = (tile % 96) << 4;
            const int n = n0 + (lane & 15), mb = m0 + ((lane >> 4) << 3);
            v8f a1 = {}; a1 = tile_gemm(h0F, HH, Wih1, HH, m0, n0, HH, lane, a1);
            const float b1 = bih1[n];
            for (int r = 0; r < 8; ++r) gIH[(mb + r) * GG + n] = (_Float16)(a1[r] + b1);
            v8f a2 = {}; a2 = tile_gemm(h1F, HH, Whh1, HH, m0, n0, HH, lane, a2);
            const float b2 = bhh1[n];
            for (int r = 0; r < 8; ++r) gHH[(mb + r) * GG + n] = (_Float16)(a2[r] + b2);
        }
        __syncthreads();
        for (int i = 0; i < 16; ++i) {          // h1 update
            const int pos = tid + 1024 * i, b = pos >> 9, j = pos & 511;
            const _Float16* gA = gIH + b * GG;
            const _Float16* gB = gHH + b * GG;
            float r  = sigm((float)gA[j] + (float)gB[j]);
            float z  = sigm((float)gA[HH + j] + (float)gB[HH + j]);
            float nn = tanhf((float)gA[2 * HH + j] + r * (float)gB[2 * HH + j]);
            float hn = (1.f - z) * nn + z * h1r[i];
            h1r[i] = hn; h1F[pos] = (_Float16)hn;
        }
        __syncthreads();
        // q = h1n@Wd^T + bd   (64 tiles / 32 waves)
        for (int tt = 0; tt < 2; ++tt) {
            const int tile = wave * 2 + tt;
            const int m0 = (tile / 32) << 4, n0 = (tile % 32) << 4;
            v8f a = {}; a = tile_gemm(h1F, HH, Wd, HH, m0, n0, HH, lane, a);
            const int n = n0 + (lane & 15), mb = m0 + ((lane >> 4) << 3);
            const float bv = bd[n];
            for (int r = 0; r < 8; ++r) qF[(mb + r) * HH + n] = (_Float16)(a[r] + bv);
        }
        __syncthreads();
        // e[b][l] = tanh(dot(enc_q[b,l,:], q[b,:]))
        for (int i = 0; i < 4; ++i) {
            const int p = tid + 1024 * i, b = p >> 7, l = p & 127;
            const _Float16* eq = encq + (size_t)(b * 128 + l) * HH;
            const _Float16* qq = qF + b * HH;
            float s = 0.f;
            for (int k = 0; k < HH; ++k) s += (float)eq[k] * (float)qq[k];
            ebuf[p] = tanhf(s);
        }
        __syncthreads();
        if (tid < 32) {                          // softmax over L per batch row
            float* e = ebuf + tid * 128;
            float mx = -1e30f;
            for (int l = 0; l < 128; ++l) mx = fmaxf(mx, e[l]);
            float s = 0.f;
            for (int l = 0; l < 128; ++l) { float v = __expf(e[l] - mx); e[l] = v; s += v; }
            const float inv = 1.f / s;
            for (int l = 0; l < 128; ++l) e[l] *= inv;
        }
        __syncthreads();
        // ctx2[b][j] = sum_l enc[b,l,j]*w[b,l]; emit feat row b*63+t
        for (int i = 0; i < 16; ++i) {
            const int pos = tid + 1024 * i, b = pos >> 9, j = pos & 511;
            const float* w = ebuf + b * 128;
            const _Float16* es = encs + (size_t)(b * 128) * HH + j;
            float s = 0.f;
            for (int l = 0; l < 128; ++l) s += w[l] * (float)es[(size_t)l * HH];
            ctxF[pos] = (_Float16)s;
            const size_t row = (size_t)(b * T1 + t) * 1024;
            feat[row + j]      = (_Float16)h1r[i];
            feat[row + HH + j] = (_Float16)s;
        }
        __syncthreads();
    }
}

// ---------------- host launch ----------------
static inline size_t al256(size_t x) { return (x + 255) & ~(size_t)255; }

extern "C" void kernel_launch(void* const* d_in, const int* in_sizes, int n_in,
                              void* d_out, int out_size, void* d_ws, size_t ws_size,
                              hipStream_t stream) {
    const int*   src_tok  = (const int*)d_in[0];
    const int*   trgt     = (const int*)d_in[1];
    const float* enc_emb  = (const float*)d_in[3];
    const float* eWih0    = (const float*)d_in[4];
    const float* eWhh0    = (const float*)d_in[5];
    const float* ebih0    = (const float*)d_in[6];
    const float* ebhh0    = (const float*)d_in[7];
    const float* eWih1    = (const float*)d_in[8];
    const float* eWhh1    = (const float*)d_in[9];
    const float* ebih1    = (const float*)d_in[10];
    const float* ebhh1    = (const float*)d_in[11];
    const float* dec_emb  = (const float*)d_in[12];
    const float* dWih0    = (const float*)d_in[13];
    const float* dWhh0    = (const float*)d_in[14];
    const float* dbih0    = (const float*)d_in[15];
    const float* dbhh0    = (const float*)d_in[16];
    const float* dWih1    = (const float*)d_in[17];
    const float* dWhh1    = (const float*)d_in[18];
    const float* dbih1    = (const float*)d_in[19];
    const float* dbhh1    = (const float*)d_in[20];
    const float* aWe      = (const float*)d_in[21];
    const float* abe      = (const float*)d_in[22];
    const float* aWd      = (const float*)d_in[23];
    const float* abd      = (const float*)d_in[24];
    const float* W1       = (const float*)d_in[25];
    const float* b1       = (const float*)d_in[26];
    const float* W2       = (const float*)d_in[27];
    const float* b2       = (const float*)d_in[28];

    // ---- workspace carve ----
    char* p = (char*)d_ws;
    auto carve = [&](size_t bytes) { char* r = p; p += al256(bytes); return r; };
    _Float16* embF    = (_Float16*)carve((size_t)4096 * EE * 2);
    float*    giEnc   = (float*)carve((size_t)2 * 4096 * GG * 4);
    _Float16* y0F     = (_Float16*)carve((size_t)4096 * 1024 * 2);
    _Float16* y1F     = (_Float16*)carve((size_t)4096 * 1024 * 2);
    _Float16* encSumF = (_Float16*)carve((size_t)4096 * HH * 2);
    _Float16* encQF   = (_Float16*)carve((size_t)4096 * HH * 2);
    _Float16* weWih0  = (_Float16*)carve((size_t)2 * GG * EE * 2);
    _Float16* weWhh0  = (_Float16*)carve((size_t)2 * GG * HH * 2);
    _Float16* weWih1  = (_Float16*)carve((size_t)2 * GG * 1024 * 2);
    _Float16* weWhh1  = (_Float16*)carve((size_t)2 * GG * HH * 2);
    _Float16* wdWih0  = (_Float16*)carve((size_t)GG * 1024 * 2);
    _Float16* wdWhh0  = (_Float16*)carve((size_t)GG * HH * 2);
    _Float16* wdWih1  = (_Float16*)carve((size_t)GG * HH * 2);
    _Float16* wdWhh1  = (_Float16*)carve((size_t)GG * HH * 2);
    _Float16* wWe     = (_Float16*)carve((size_t)HH * HH * 2);
    _Float16* wWd     = (_Float16*)carve((size_t)HH * HH * 2);
    _Float16* wW1     = (_Float16*)carve((size_t)1024 * 1024 * 2);
    _Float16* wW2     = (_Float16*)carve((size_t)VV * 1024 * 2);
    _Float16* dEmbF   = (_Float16*)carve((size_t)2016 * EE * 2);
    float*    giDec   = (float*)carve((size_t)2016 * GG * 4);
    _Float16* featF   = (_Float16*)carve((size_t)2016 * 1024 * 2);
    _Float16* hidF    = (_Float16*)carve((size_t)2016 * 1024 * 2);

    // ---- weight conversions f32 -> f16 ----
    auto cvt = [&](const float* s, _Float16* d, size_t n) {
        cvt_f32_f16<<<dim3((unsigned)((n + 255) / 256)), 256, 0, stream>>>(s, d, (int)n);
    };
    cvt(eWih0, weWih0, (size_t)2 * GG * EE);
    cvt(eWhh0, weWhh0, (size_t)2 * GG * HH);
    cvt(eWih1, weWih1, (size_t)2 * GG * 1024);
    cvt(eWhh1, weWhh1, (size_t)2 * GG * HH);
    cvt(dWih0, wdWih0, (size_t)GG * 1024);
    cvt(dWhh0, wdWhh0, (size_t)GG * HH);
    cvt(dWih1, wdWih1, (size_t)GG * HH);
    cvt(dWhh1, wdWhh1, (size_t)GG * HH);
    cvt(aWe, wWe, (size_t)HH * HH);
    cvt(aWd, wWd, (size_t)HH * HH);
    cvt(W1, wW1, (size_t)1024 * 1024);
    cvt(W2, wW2, (size_t)VV * 1024);

    auto gemmF32 = [&](const _Float16* A, const _Float16* W, const float* bias, float* C,
                       int M, int N, int K, int lda, int ldb, int ldc) {
        int blocks = ((M >> 5) * (N >> 6) + 7) / 8;
        gemm_nt<0, 0><<<dim3(blocks), 256, 0, stream>>>(A, W, bias, C, M, N, K, lda, ldb, ldc);
    };
    auto gemmF16 = [&](const _Float16* A, const _Float16* W, const float* bias, _Float16* C,
                       int M, int N, int K, int lda, int ldb, int ldc, bool relu) {
        int blocks = ((M >> 5) * (N >> 6) + 7) / 8;
        if (relu) gemm_nt<1, 1><<<dim3(blocks), 256, 0, stream>>>(A, W, bias, C, M, N, K, lda, ldb, ldc);
        else      gemm_nt<1, 0><<<dim3(blocks), 256, 0, stream>>>(A, W, bias, C, M, N, K, lda, ldb, ldc);
    };

    // ---- encoder ----
    gather_enc<<<dim3(4096), 256, 0, stream>>>(src_tok, enc_emb, embF);
    for (int d = 0; d < 2; ++d)                               // layer-0 input GEMMs
        gemmF32(embF, weWih0 + (size_t)d * GG * EE, ebih0 + d * GG,
                giEnc + (size_t)d * 4096 * GG, 4096, GG, EE, EE, EE, GG);
    size_t ldsEnc = (size_t)32 * HH * 2 + (size_t)32 * GG * 4;   // 229376 B
    gru_enc<<<dim3(2), 1024, ldsEnc, stream>>>(giEnc, weWhh0, ebhh0, y0F);
    for (int d = 0; d < 2; ++d)                               // layer-1 input GEMMs
        gemmF32(y0F, weWih1 + (size_t)d * GG * 1024, ebih1 + d * GG,
                giEnc + (size_t)d * 4096 * GG, 4096, GG, 1024, 1024, 1024, GG);
    gru_enc<<<dim3(2), 1024, ldsEnc, stream>>>(giEnc, weWhh1, ebhh1, y1F);
    enc_sum_tr<<<dim3(4096), 256, 0, stream>>>(y1F, encSumF);
    gemmF16(encSumF, wWe, abe, encQF, 4096, HH, HH, HH, HH, HH, false);

    // ---- decoder recurrence ----
    gather_dec<<<dim3(2016), 256, 0, stream>>>(trgt, dec_emb, dEmbF);
    gemmF32(dEmbF, wdWih0, dbih0, giDec, 2016, GG, EE, EE, 1024, GG);
    size_t ldsDec = ((size_t)2 * 32 * GG + (size_t)3 * 32 * HH) * 2;  // 294912 B
    gru_dec<<<dim3(1), 1024, ldsDec, stream>>>(giDec, wdWih0 + HH, wdWhh0, dbhh0,
                                               wdWih1, dbih1, wdWhh1, dbhh1,
                                               wWd, abd, encQF, encSumF, featF);

    // ---- output MLP (bulk of the FLOPs, full-chip WMMA GEMMs) ----
    gemmF16(featF, wW1, b1, hidF, 2016, 1024, 1024, 1024, 1024, 1024, true);
    gemmF32(hidF, wW2, b2, (float*)d_out, 2016, VV, 1024, 1024, 1024, VV);
}
